// FFTCNN_15934328668623
// MI455X (gfx1250) — matrix operand order repaired and checked by reading, not compiled, not run on hardware
//
#include <hip/hip_runtime.h>

// ---- types -----------------------------------------------------------------
typedef __attribute__((ext_vector_type(16))) __bf16 v16bf;
typedef __attribute__((ext_vector_type(8)))  float  v8f;
typedef unsigned short u16;
typedef unsigned int   u32;

#define BATCH 32
#define LLEN  4096
#define EMB   512
#define HID   128

__device__ __forceinline__ u16 f2bf(float f) {
    u32 u = __float_as_uint(f);
    u32 r = u + 0x7FFFu + ((u >> 16) & 1u);   // round-to-nearest-even
    return (u16)(r >> 16);
}

// A-fragment K mapping for 16-bit A 16x32 (contiguous, matches B loads):
// lanes 0-15: e0..15 -> K=0..15
// lanes16-31: e0..15 -> K=16..31
__device__ __forceinline__ int a_kmap(int lane, int e) {
    return (lane < 16) ? e : e + 16;
}

// ---- kernel 1: pack W1/W2 (fp32 [O][C][3]) into bf16 WMMA-A fragments ------
// A1: [mtile(8)][kstep(48)][lane(32)][e(16)]  kk = 32*s + klocal, tap = s/16
// A2: [mtile(8)][kstep(12)][lane(32)][e(16)]  tap = s/4
__global__ void FFTCNN_pack_w(const float* __restrict__ w1,
                              const float* __restrict__ w2,
                              u16* __restrict__ a1, u16* __restrict__ a2) {
    const int N1 = 8 * 48 * 32 * 16;
    const int N2 = 8 * 12 * 32 * 16;
    int idx = blockIdx.x * blockDim.x + threadIdx.x;
    if (idx < N1) {
        int e = idx & 15, l = (idx >> 4) & 31;
        int s = (idx >> 9) % 48, t = idx / (512 * 48);
        int oc  = t * 16 + (l & 15);
        int tap = s >> 4;
        int c   = ((s & 15) << 5) + a_kmap(l, e);
        a1[idx] = f2bf(w1[(oc * EMB + c) * 3 + tap]);
    } else if (idx < N1 + N2) {
        int j = idx - N1;
        int e = j & 15, l = (j >> 4) & 31;
        int s = (j >> 9) % 12, t = j / (512 * 12);
        int oc  = t * 16 + (l & 15);
        int tap = s >> 2;
        int c   = ((s & 3) << 5) + a_kmap(l, e);
        a2[j] = f2bf(w2[(oc * HID + c) * 3 + tap]);
    }
}

// ---- kernel 2: embedding gather -> bf16 activations [B][L][EMB] ------------
__global__ void FFTCNN_embed(const int* __restrict__ tokens,
                             const float* __restrict__ emb,
                             u16* __restrict__ xb) {
    int p   = blockIdx.x;              // b*L + l
    int tok = tokens[p];
    int c   = threadIdx.x * 2;         // 256 threads x 2 channels
    float2 v = *reinterpret_cast<const float2*>(emb + (size_t)tok * EMB + c);
    ushort2 o;
    o.x = f2bf(v.x);
    o.y = f2bf(v.y);
    *reinterpret_cast<ushort2*>(xb + (size_t)p * EMB + c) = o;
}

__global__ void FFTCNN_init_pool(float* __restrict__ pooled) {
    int i = blockIdx.x * blockDim.x + threadIdx.x;
    if (i < BATCH * HID) pooled[i] = 0.0f;
}

// ---- kernel 3: conv1 (circular, 3-tap) as WMMA GEMM + bias + ReLU ----------
// block = 8 waves; wave w owns out-channel tile w; block covers 64 positions.
// All B addresses derive from 12 precomputed base pointers + immediate offsets.
__global__ __launch_bounds__(256) void FFTCNN_conv1(
        const u16* __restrict__ xb, const u16* __restrict__ a1,
        const float* __restrict__ b1, u16* __restrict__ y1) {
    const int lane = threadIdx.x & 31;
    const int w    = threadIdx.x >> 5;
    const int b    = blockIdx.x >> 6;          // 64 blocks per batch
    const int l0   = (blockIdx.x & 63) * 64;
    const int n    = lane & 15;
    const int koff = (lane >> 4) << 4;         // B frag: upper lanes take K+16

    const __bf16* x  = reinterpret_cast<const __bf16*>(xb) + (size_t)b * LLEN * EMB;
    const __bf16* ap = reinterpret_cast<const __bf16*>(a1) + (size_t)w * 48 * 512 + lane * 16;

    // 12 base pointers: circular wrap resolved once, outside the K-loop
    const __bf16* bp[3][4];
#pragma unroll
    for (int tap = 0; tap < 3; ++tap)
#pragma unroll
        for (int nt = 0; nt < 4; ++nt) {
            int pos = (l0 + nt * 16 + n - tap + LLEN) & (LLEN - 1);
            bp[tap][nt] = x + (size_t)pos * EMB + koff;
        }

    v8f acc[4] = {};
#pragma unroll
    for (int s = 0; s < 48; ++s) {             // full unroll: offsets become imms
        v16bf a = *reinterpret_cast<const v16bf*>(ap + s * 512);
        const int tap = s >> 4;
        const int c0  = (s & 15) << 5;
#pragma unroll
        for (int nt = 0; nt < 4; ++nt) {
            v16bf bv = *reinterpret_cast<const v16bf*>(bp[tap][nt] + c0);
            acc[nt] = __builtin_amdgcn_wmma_f32_16x16x32_bf16(
                false, a, false, bv, (short)0, acc[nt], false, false);
        }
    }
    // epilogue: bias + ReLU, store bf16 [B][L][HID]
    const int mbase = (lane >> 4) * 8;
#pragma unroll
    for (int r = 0; r < 8; ++r) {
        int oc = w * 16 + mbase + r;
        float bias = b1[oc];
#pragma unroll
        for (int nt = 0; nt < 4; ++nt) {
            float v = acc[nt][r] + bias;
            v = v > 0.0f ? v : 0.0f;
            int pos = l0 + nt * 16 + n;
            y1[((size_t)b * LLEN + pos) * HID + oc] = f2bf(v);
        }
    }
}

// ---- kernel 4: conv2 + bias + ReLU + fused global max-pool -----------------
__global__ __launch_bounds__(256) void FFTCNN_conv2(
        const u16* __restrict__ y1b, const u16* __restrict__ a2,
        const float* __restrict__ b2, float* __restrict__ pooled) {
    const int lane = threadIdx.x & 31;
    const int w    = threadIdx.x >> 5;
    const int b    = blockIdx.x >> 6;
    const int l0   = (blockIdx.x & 63) * 64;
    const int n    = lane & 15;
    const int koff = (lane >> 4) << 4;

    const __bf16* x  = reinterpret_cast<const __bf16*>(y1b) + (size_t)b * LLEN * HID;
    const __bf16* ap = reinterpret_cast<const __bf16*>(a2) + (size_t)w * 12 * 512 + lane * 16;

    const __bf16* bp[3][4];
#pragma unroll
    for (int tap = 0; tap < 3; ++tap)
#pragma unroll
        for (int nt = 0; nt < 4; ++nt) {
            int pos = (l0 + nt * 16 + n - tap + LLEN) & (LLEN - 1);
            bp[tap][nt] = x + (size_t)pos * HID + koff;
        }

    v8f acc[4] = {};
#pragma unroll
    for (int s = 0; s < 12; ++s) {
        v16bf a = *reinterpret_cast<const v16bf*>(ap + s * 512);
        const int tap = s >> 2;
        const int c0  = (s & 3) << 5;
#pragma unroll
        for (int nt = 0; nt < 4; ++nt) {
            v16bf bv = *reinterpret_cast<const v16bf*>(bp[tap][nt] + c0);
            acc[nt] = __builtin_amdgcn_wmma_f32_16x16x32_bf16(
                false, a, false, bv, (short)0, acc[nt], false, false);
        }
    }
    // epilogue: bias + ReLU, max over this block's 64 positions, atomic-max.
    const int mbase = (lane >> 4) * 8;
#pragma unroll
    for (int r = 0; r < 8; ++r) {
        float bias = b2[w * 16 + mbase + r];
        float m = 0.0f;                       // ReLU floor == identity for max
#pragma unroll
        for (int nt = 0; nt < 4; ++nt) {
            float v = acc[nt][r] + bias;
            m = v > m ? v : m;
        }
        // reduce across the 16 lanes of each half-wave (rows stay separate)
        for (int off = 1; off < 16; off <<= 1)
            m = fmaxf(m, __shfl_xor(m, off, 32));
        if (n == 0) {                         // lanes 0 and 16
            // non-negative floats: uint bit-pattern compare == float compare
            atomicMax(reinterpret_cast<u32*>(&pooled[b * HID + w * 16 + mbase + r]),
                      __float_as_uint(m));
        }
    }
}

// ---- kernel 5: tiny MLP head (fp32) ---------------------------------------
__global__ void FFTCNN_head(const float* __restrict__ pooled,
                            const float* __restrict__ lw1, const float* __restrict__ lb1,
                            const float* __restrict__ lw2, const float* __restrict__ lb2,
                            float* __restrict__ out) {
    __shared__ float sp[BATCH * HID];
    __shared__ float sh[BATCH * HID];
    for (int i = threadIdx.x; i < BATCH * HID; i += 256) sp[i] = pooled[i];
    __syncthreads();
    for (int i = threadIdx.x; i < BATCH * HID; i += 256) {
        int b = i >> 7, j = i & 127;
        float s = lb1[j];
        for (int c = 0; c < HID; ++c) s += sp[b * HID + c] * lw1[j * HID + c];
        sh[i] = s > 0.0f ? s : 0.0f;
    }
    __syncthreads();
    for (int i = threadIdx.x; i < BATCH * 6; i += 256) {
        int b = i / 6, c = i % 6;
        float s = lb2[c];
        for (int j = 0; j < HID; ++j) s += sh[b * HID + j] * lw2[c * HID + j];
        out[i] = s;
    }
}

// ---- launcher --------------------------------------------------------------
extern "C" void kernel_launch(void* const* d_in, const int* in_sizes, int n_in,
                              void* d_out, int out_size, void* d_ws, size_t ws_size,
                              hipStream_t stream) {
    const int*   tokens = (const int*)  d_in[0];
    const float* emb    = (const float*)d_in[1];
    const float* w1     = (const float*)d_in[2];
    const float* b1     = (const float*)d_in[3];
    const float* w2     = (const float*)d_in[4];
    const float* b2     = (const float*)d_in[5];
    const float* lw1    = (const float*)d_in[6];
    const float* lb1    = (const float*)d_in[7];
    const float* lw2    = (const float*)d_in[8];
    const float* lb2    = (const float*)d_in[9];

    char* ws = (char*)d_ws;
    u16*   a1     = (u16*)  (ws + 0);                        // 384 KB
    u16*   a2     = (u16*)  (ws + 393216);                   //  96 KB
    float* pooled = (float*)(ws + 491520);                   //  16 KB
    u16*   xb     = (u16*)  (ws + 524288);                   // 128 MB
    u16*   y1     = (u16*)  (ws + 524288 + (size_t)BATCH * LLEN * EMB * 2); // 32 MB
    float* out    = (float*)d_out;

    const int NPACK = 8 * 48 * 32 * 16 + 8 * 12 * 32 * 16;   // 245760
    FFTCNN_pack_w   <<<(NPACK + 255) / 256, 256, 0, stream>>>(w1, w2, a1, a2);
    FFTCNN_embed    <<<BATCH * LLEN, 256, 0, stream>>>(tokens, emb, xb);
    FFTCNN_init_pool<<<(BATCH * HID + 255) / 256, 256, 0, stream>>>(pooled);
    FFTCNN_conv1    <<<BATCH * (LLEN / 64), 256, 0, stream>>>(xb, a1, b1, y1);
    FFTCNN_conv2    <<<BATCH * (LLEN / 64), 256, 0, stream>>>(y1, a2, b2, pooled);
    FFTCNN_head     <<<1, 256, 0, stream>>>(pooled, lw1, lb1, lw2, lb2, out);
}
